// MetaVisualLearner_541165879792
// MI455X (gfx1250) — matrix operand order repaired and checked by reading, not compiled, not run on hardware
//
#include <hip/hip_runtime.h>
#include <hip/hip_bf16.h>

typedef __attribute__((ext_vector_type(16))) _Float16 v16h;
typedef __attribute__((ext_vector_type(8)))  _Float16 v8h;
typedef __attribute__((ext_vector_type(4)))  _Float16 v4h;
typedef __attribute__((ext_vector_type(8)))  float    v8f;

#define BB_   2
#define NN_   4096
#define KK_   16
#define DD_   128
#define MM_   2
#define HID_  128
#define KEYD_ 64
#define NKTOT (NN_*KK_)      // 65536
#define ETOT  (BB_*NN_*KK_)  // 131072
#define TAU_  0.2f

#define EDGE_STR 264   // 256 cols + 8 pad (f16), keeps 16B alignment per row
#define H_STR    136   // 128 cols + 8 pad
#define WAVES    2     // waves per block; each wave owns TWO 16-row tiles

// packed-weight offsets in f16 elements inside d_ws
#define OFF_ENC0 0u          // 256x128
#define OFF_ENC1 32768u      // 128x128
#define OFF_ENC2 49152u      // 128x128
#define OFF_ENC3 65536u      // 128x64
#define OFF_B0   73728u      // 2 x 256x128
#define OFF_B1   139264u     // 2 x 128x128
#define OFF_B2   172032u     // 2 x 128x128
// total = 204800 f16 = 409600 bytes in d_ws

// ---------------------------------------------------------------------------
// Repack f32 row-major weight W[Kd][Nd] into WMMA f16 B-fragment layout:
// fragment (kc,nt): lane l holds col n = nt*16+(l&15), k = kc*32+((l>>4)<<4)+j
// stored as 16 contiguous f16 per lane -> 32B coalesced per-lane loads.
// ---------------------------------------------------------------------------
__global__ void pack_w(const float* __restrict__ src, _Float16* __restrict__ dst,
                       int Kd, int Nd) {
    int t = blockIdx.x * blockDim.x + threadIdx.x;
    int nNT = Nd >> 4;
    int nFrag = (Kd >> 5) * nNT;
    int frag = t >> 5, lane = t & 31;
    if (frag >= nFrag) return;
    int kc = frag / nNT, nt = frag - kc * nNT;
    int n = nt * 16 + (lane & 15);
    int kbase = kc * 32 + ((lane >> 4) << 4);
    _Float16* o = dst + ((size_t)frag * 32 + lane) * 16;
#pragma unroll
    for (int j = 0; j < 16; ++j)
        o[j] = (_Float16)src[(size_t)(kbase + j) * Nd + n];
}

// ---------------------------------------------------------------------------
// One MLP layer applied to TWO 16-row tiles sharing every B fragment:
//   out{0,1}[16xND] = act(in{0,1}[16xKD] * W + b)
// A fragments from wave-private LDS; B fragments (pre-packed f16) from global,
// loaded once and fed to two independent WMMAs (2x weight-bandwidth reuse).
// ACT: 0 = none, 1 = ReLU, 2 = exact-erf GELU
// ---------------------------------------------------------------------------
template<int KD, int ND, int ACT>
__device__ __forceinline__ void mlp_layer2(const _Float16* in0, const _Float16* in1,
                                           int inStride,
                                           const _Float16* __restrict__ wf,
                                           const float* __restrict__ bias,
                                           _Float16* out0, _Float16* out1,
                                           int outStride, int lane)
{
    constexpr int NKC = KD / 32, NNT = ND / 16;
    const int col  = lane & 15;            // also the A-row this lane serves
    const int mrow = (lane >> 4) << 3;     // C-layout row base
    const size_t rowoff = (size_t)col * inStride + ((lane >> 4) << 4);
    const _Float16* rp0 = in0 + rowoff;
    const _Float16* rp1 = in1 + rowoff;

    v16h a0[NKC], a1[NKC];
#pragma unroll
    for (int kc = 0; kc < NKC; ++kc) {
        v8h lo0 = *(const v8h*)(rp0 + kc * 32);
        v8h hi0 = *(const v8h*)(rp0 + kc * 32 + 8);
        v8h lo1 = *(const v8h*)(rp1 + kc * 32);
        v8h hi1 = *(const v8h*)(rp1 + kc * 32 + 8);
#pragma unroll
        for (int j = 0; j < 8; ++j) {
            a0[kc][j] = lo0[j]; a0[kc][j + 8] = hi0[j];
            a1[kc][j] = lo1[j]; a1[kc][j + 8] = hi1[j];
        }
    }

#pragma unroll
    for (int nt = 0; nt < NNT; ++nt) {
        float bv = bias[nt * 16 + col];
        v8f c0, c1;
#pragma unroll
        for (int r = 0; r < 8; ++r) { c0[r] = bv; c1[r] = bv; }
#pragma unroll
        for (int kc = 0; kc < NKC; ++kc) {
            v16h bf = *(const v16h*)(wf + (size_t)(((kc * NNT + nt) << 5) + lane) * 16);
            c0 = __builtin_amdgcn_wmma_f32_16x16x32_f16(
                     false, a0[kc], false, bf, (short)0, c0, false, false);
            c1 = __builtin_amdgcn_wmma_f32_16x16x32_f16(
                     false, a1[kc], false, bf, (short)0, c1, false, false);
        }
#pragma unroll
        for (int r = 0; r < 8; ++r) {
            float v0 = c0[r], v1 = c1[r];
            if (ACT == 1) {
                v0 = fmaxf(v0, 0.0f); v1 = fmaxf(v1, 0.0f);
            } else if (ACT == 2) {
                v0 = 0.5f * v0 * (1.0f + erff(v0 * 0.70710678118654752f));
                v1 = 0.5f * v1 * (1.0f + erff(v1 * 0.70710678118654752f));
            }
            out0[(size_t)(mrow + r) * outStride + nt * 16 + col] = (_Float16)v0;
            out1[(size_t)(mrow + r) * outStride + nt * 16 + col] = (_Float16)v1;
        }
    }
}

// ---------------------------------------------------------------------------
// Fused: gather edge feats -> encoder MLP -> cosine attention
//        -> 2x bias MLPs -> combine with affinities.
// 2 waves/block, each wave owns 32 edges = two 16-row tiles (wave-private
// LDS, no barriers; B fragments reused across both tiles).
// ---------------------------------------------------------------------------
__global__ __launch_bounds__(64) void fused_mlp_attn(
    const float* __restrict__ bbf, const int* __restrict__ xidx, const int* __restrict__ yidx,
    const float* __restrict__ aff, const float* __restrict__ emb, const int* __restrict__ kid,
    const float* __restrict__ encb0, const float* __restrict__ encb1,
    const float* __restrict__ encb2, const float* __restrict__ encb3,
    const float* __restrict__ bb0, const float* __restrict__ bb1,
    const float* __restrict__ bb2, const float* __restrict__ bb3,
    const float* __restrict__ bw3,
    const _Float16* __restrict__ wp, float* __restrict__ out)
{
    __shared__ __align__(16) _Float16 s_edge[WAVES][32][EDGE_STR]; // 33792 B
    __shared__ __align__(16) _Float16 s_h[WAVES][32][H_STR];       // 17408 B

    const int lane = threadIdx.x & 31;
    const int wave = threadIdx.x >> 5;
    const long tileBase = ((long)blockIdx.x * WAVES + wave) * 32;  // 32 edges/wave

    _Float16 (*edge)[EDGE_STR] = s_edge[wave];
    _Float16 (*hbuf)[H_STR]    = s_h[wave];
    _Float16 *ep0 = &edge[0][0],  *ep1 = &edge[16][0];
    _Float16 *hp0 = &hbuf[0][0],  *hp1 = &hbuf[16][0];

    // --- 1) gather neighbor features, concat [xf|yf], convert to f16 in LDS
    const int c4 = lane << 2;   // 4 floats per lane -> 128 cols / 32 lanes
#pragma unroll 4
    for (int i = 0; i < 32; ++i) {
        long e = tileBase + i;
        int b  = (int)(e >> 16);                 // e / (N*K)
        int xi = xidx[e], yi = yidx[e];
        const float4 xv = *(const float4*)(bbf + (((size_t)b * NN_ + xi) << 7) + c4);
        const float4 yv = *(const float4*)(bbf + (((size_t)b * NN_ + yi) << 7) + c4);
        v4h hx = { (_Float16)xv.x, (_Float16)xv.y, (_Float16)xv.z, (_Float16)xv.w };
        v4h hy = { (_Float16)yv.x, (_Float16)yv.y, (_Float16)yv.z, (_Float16)yv.w };
        *(v4h*)&edge[i][c4]       = hx;
        *(v4h*)&edge[i][DD_ + c4] = hy;
    }

    // --- 2) encoder MLP (ReLU x3, linear out 64) -> cond in hbuf cols [0,64)
    mlp_layer2<256, 128, 1>(ep0, ep1, EDGE_STR, wp + OFF_ENC0, encb0, hp0, hp1, H_STR, lane);
    mlp_layer2<128, 128, 1>(hp0, hp1, H_STR,    wp + OFF_ENC1, encb1, hp0, hp1, H_STR, lane);
    mlp_layer2<128, 128, 1>(hp0, hp1, H_STR,    wp + OFF_ENC2, encb2, hp0, hp1, H_STR, lane);
    mlp_layer2<128,  64, 0>(hp0, hp1, H_STR,    wp + OFF_ENC3, encb3, hp0, hp1, H_STR, lane);

    // --- 3) cosine attention per tile: lane (row, half) covers 32 key dims
    const int row = lane & 15, half = lane >> 4;
    const float* e0 = emb + (size_t)kid[0] * KEYD_;
    const float* e1 = emb + (size_t)kid[1] * KEYD_;
    float attn_m0[2], attn_m1[2];
#pragma unroll
    for (int t = 0; t < 2; ++t) {
        float sumsq = 0.f, dot0 = 0.f, dot1 = 0.f, g0s = 0.f, g1s = 0.f;
#pragma unroll
        for (int c = 0; c < 32; ++c) {
            int cc = half * 32 + c;
            float cv = (float)hbuf[t * 16 + row][cc];
            float g0 = e0[cc], g1 = e1[cc];
            sumsq += cv * cv; dot0 += cv * g0; dot1 += cv * g1;
            g0s += g0 * g0;   g1s += g1 * g1;
        }
        sumsq += __shfl_xor(sumsq, 16);
        dot0  += __shfl_xor(dot0, 16);
        dot1  += __shfl_xor(dot1, 16);
        g0s   += __shfl_xor(g0s, 16);
        g1s   += __shfl_xor(g1s, 16);
        float cn = fmaxf(sqrtf(sumsq), 1e-12f);
        float z0 = (dot0 / (cn * fmaxf(sqrtf(g0s), 1e-12f))) * (1.0f / TAU_);
        float z1 = (dot1 / (cn * fmaxf(sqrtf(g1s), 1e-12f))) * (1.0f / TAU_);
        attn_m0[t] = 1.0f / (1.0f + expf(-z0));
        attn_m1[t] = 1.0f / (1.0f + expf(-z1));
    }

    // --- 4) per-component bias MLPs (GELU x3, final 128->1 in VALU)
    float ebias[MM_][2];
#pragma unroll
    for (int m = 0; m < MM_; ++m) {
        const _Float16* w0 = wp + OFF_B0 + (size_t)m * 32768;
        const _Float16* w1 = wp + OFF_B1 + (size_t)m * 16384;
        const _Float16* w2 = wp + OFF_B2 + (size_t)m * 16384;
        mlp_layer2<256, 128, 2>(ep0, ep1, EDGE_STR, w0, bb0 + m * HID_, hp0, hp1, H_STR, lane);
        mlp_layer2<128, 128, 2>(hp0, hp1, H_STR,    w1, bb1 + m * HID_, hp0, hp1, H_STR, lane);
        mlp_layer2<128, 128, 2>(hp0, hp1, H_STR,    w2, bb2 + m * HID_, hp0, hp1, H_STR, lane);
        const float* w3 = bw3 + (size_t)m * HID_;   // bias_W3[m,128,1]
#pragma unroll
        for (int t = 0; t < 2; ++t) {
            float s = 0.f;
#pragma unroll
            for (int c = 0; c < 64; ++c) {
                int cc = half * 64 + c;
                s += (float)hbuf[t * 16 + row][cc] * w3[cc];
            }
            s += __shfl_xor(s, 16);
            ebias[m][t] = s + bb3[m];
        }
    }

    // --- 5) combine: out = sum_m attn_m * (aff_m - bias_m)
    //     lanes 0-15 store tile0 rows, lanes 16-31 store tile1 rows (no diverge)
    {
        int  t = half;
        long e = tileBase + t * 16 + row;
        int  b = (int)(e >> 16);
        long r = e & (NKTOT - 1);
        const float* affb = aff + ((size_t)b * MM_) * NKTOT + r;
        float a0v = t ? attn_m0[1] : attn_m0[0];
        float a1v = t ? attn_m1[1] : attn_m1[0];
        float b0v = t ? ebias[0][1] : ebias[0][0];
        float b1v = t ? ebias[1][1] : ebias[1][0];
        out[e] = a0v * (affb[0]     - b0v)
               + a1v * (affb[NKTOT] - b1v);
    }
}

extern "C" void kernel_launch(void* const* d_in, const int* in_sizes, int n_in,
                              void* d_out, int out_size, void* d_ws, size_t ws_size,
                              hipStream_t stream) {
    (void)in_sizes; (void)n_in; (void)out_size; (void)ws_size;
    const float* bbf  = (const float*)d_in[0];
    const int*   xidx = (const int*)d_in[1];
    const int*   yidx = (const int*)d_in[2];
    const float* aff  = (const float*)d_in[3];
    const float* emb  = (const float*)d_in[4];
    const int*   kid  = (const int*)d_in[5];
    const float* encW0 = (const float*)d_in[6];  const float* encb0 = (const float*)d_in[7];
    const float* encW1 = (const float*)d_in[8];  const float* encb1 = (const float*)d_in[9];
    const float* encW2 = (const float*)d_in[10]; const float* encb2 = (const float*)d_in[11];
    const float* encW3 = (const float*)d_in[12]; const float* encb3 = (const float*)d_in[13];
    const float* bW0 = (const float*)d_in[14]; const float* bb0 = (const float*)d_in[15];
    const float* bW1 = (const float*)d_in[16]; const float* bb1 = (const float*)d_in[17];
    const float* bW2 = (const float*)d_in[18]; const float* bb2 = (const float*)d_in[19];
    const float* bW3 = (const float*)d_in[20]; const float* bb3 = (const float*)d_in[21];

    _Float16* wp = (_Float16*)d_ws;
    auto pk = [&](const float* s, unsigned off, int Kd, int Nd) {
        int th = (Kd / 32) * (Nd / 16) * 32;
        pack_w<<<(th + 255) / 256, 256, 0, stream>>>(s, wp + off, Kd, Nd);
    };
    pk(encW0, OFF_ENC0, 256, 128);
    pk(encW1, OFF_ENC1, 128, 128);
    pk(encW2, OFF_ENC2, 128, 128);
    pk(encW3, OFF_ENC3, 128, 64);
    pk(bW0,         OFF_B0,         256, 128);
    pk(bW0 + 32768, OFF_B0 + 32768, 256, 128);
    pk(bW1,         OFF_B1,         128, 128);
    pk(bW1 + 16384, OFF_B1 + 16384, 128, 128);
    pk(bW2,         OFF_B2,         128, 128);
    pk(bW2 + 16384, OFF_B2 + 16384, 128, 128);

    fused_mlp_attn<<<ETOT / (32 * WAVES), 32 * WAVES, 0, stream>>>(
        bbf, xidx, yidx, aff, emb, kid,
        encb0, encb1, encb2, encb3,
        bb0, bb1, bb2, bb3, bW3,
        wp, (float*)d_out);
}
